// Transducer_58832462020720
// MI455X (gfx1250) — matrix operand order repaired and checked by reading, not compiled, not run on hardware
//
#include <hip/hip_runtime.h>
#include <hip/hip_bf16.h>

// ---------------- problem constants ----------------
#define Bz     4
#define Tz     200
#define Uz     100      // tokens; dec rows = Uz+1 = 101
#define Hz     320
#define INNERz 512
#define Vz     1000
#define VP     1008     // V padded to 63 tiles of 16
#define UP     112      // u rows padded to 7 tiles of 16
#define LSTR   112      // row stride of blank/label arrays
#define BSTR   520      // f16 stride of staged B rows in LDS (conflict-free)

typedef _Float16 h16;
typedef __attribute__((ext_vector_type(16))) _Float16 v16h;
typedef __attribute__((ext_vector_type(8)))  _Float16 v8h;
typedef __attribute__((ext_vector_type(8)))  float    v8f;

// hardware tanh if this toolchain exposes it (gfx1250 v_tanh_f32), else ocml
#if __has_builtin(__builtin_amdgcn_tanhf)
  #define FAST_TANH(x) __builtin_amdgcn_tanhf(x)
#elif __has_builtin(__builtin_amdgcn_tanh_f32)
  #define FAST_TANH(x) __builtin_amdgcn_tanh_f32(x)
#else
  #define FAST_TANH(x) tanhf(x)
#endif

#if __has_builtin(__builtin_amdgcn_s_wait_asynccnt)
  #define WAIT_ASYNC0() __builtin_amdgcn_s_wait_asynccnt(0)
#else
  #define WAIT_ASYNC0() asm volatile("s_wait_asynccnt 0x0" ::: "memory")
#endif

// ---------------- kernel 0: weight prep ----------------
// W2T[vp][k] = (f16) W2[k][vp]  (zero-pad vp>=1000);  b2p padded with -inf
__global__ __launch_bounds__(256)
void prep_kernel(const float* __restrict__ W2, const float* __restrict__ b2,
                 h16* __restrict__ W2T, float* __restrict__ b2p)
{
    int idx = blockIdx.x * 256 + threadIdx.x;
    if (idx < VP * INNERz) {
        int vp = idx >> 9;          // /512
        int k  = idx & 511;
        float w = (vp < Vz) ? W2[(size_t)k * Vz + vp] : 0.0f;
        W2T[(size_t)vp * INNERz + k] = (h16)w;
    }
    if (idx < VP) b2p[idx] = (idx < Vz) ? b2[idx] : -__builtin_inff();
}

// ---------------- kernel 1: h_enc' = enc@W1[:H]+b1 ; h_dec' = dec@W1[H:] ----------------
__global__ __launch_bounds__(256)
void row_gemm_kernel(const float* __restrict__ enc, const float* __restrict__ dec,
                     const float* __restrict__ W1, const float* __restrict__ b1,
                     float* __restrict__ henc, float* __restrict__ hdec)
{
    __shared__ float xs[Hz];
    int row = blockIdx.x;
    const float* x; const float* W; float* out; bool addb;
    if (row < Bz * Tz) {
        x = enc + (size_t)row * Hz;  W = W1;
        out = henc + (size_t)row * INNERz;  addb = true;
    } else {
        int r = row - Bz * Tz;
        x = dec + (size_t)r * Hz;  W = W1 + (size_t)Hz * INNERz;
        out = hdec + (size_t)r * INNERz;  addb = false;
    }
    for (int i = threadIdx.x; i < Hz; i += 256) xs[i] = x[i];
    __syncthreads();
    for (int i = threadIdx.x; i < INNERz; i += 256) {
        float acc = addb ? b1[i] : 0.0f;
        #pragma unroll 4
        for (int h = 0; h < Hz; h++) acc = fmaf(xs[h], W[(size_t)h * INNERz + i], acc);
        out[i] = acc;
    }
}

// ---------------- kernel 2: fused joint + log-softmax (WMMA) ----------------
// Async copy of one 16x512 f16 B tile (16 KB) global -> LDS, BSTR-padded rows.
// Uses GLOBAL_LOAD_ASYNC_TO_LDS_B128 (GVS mode: lds_vdst, voffset32, sbase64),
// tracked by ASYNCcnt; no VGPR data round-trip.
__device__ __forceinline__ void stage_tile_async(int tid, const h16* __restrict__ W2T,
                                                 int vt, h16* dstbuf)
{
    uint32_t ldsbase = (uint32_t)(uintptr_t)dstbuf;   // low 32 bits of generic = LDS offset
    uint32_t gbase   = (uint32_t)vt * (16u * INNERz * 2u);   // tile byte offset
    #pragma unroll 1
    for (int i = tid; i < 1024; i += 224) {           // 1024 x 16B lane-transfers
        // row = i>>6 (64 x 16B per 1KB source row); LDS row stride = 1040B
        uint32_t lds = ldsbase + (uint32_t)i * 16u + (uint32_t)(i >> 6) * 16u;
        uint32_t gof = gbase + (uint32_t)i * 16u;
        asm volatile("global_load_async_to_lds_b128 %0, %1, %2"
                     :: "v"(lds), "v"(gof), "s"(W2T) : "memory");
    }
}

__global__ __launch_bounds__(224)
void joint_kernel(const float* __restrict__ henc, const float* __restrict__ hdec,
                  const h16* __restrict__ W2T, const float* __restrict__ b2p,
                  const int* __restrict__ tokens,
                  float* __restrict__ blankO, float* __restrict__ labelO)
{
    __shared__ __align__(16) h16 Bst[2][16 * BSTR];
    __shared__ float henc_s[INNERz];
    __shared__ float lse_s[UP];

    const int tid  = threadIdx.x;
    const int wave = tid >> 5;
    const int lane = tid & 31;
    const int m    = lane & 15;   // row-in-tile / column-in-tile
    const int hi   = lane >> 4;   // half-wave id
    const int bt   = blockIdx.x;
    const int b    = bt / Tz, t = bt % Tz;
    const int u    = wave * 16 + m;       // this lane's u row (A-matrix M index)

    for (int i = tid; i < INNERz; i += 224) henc_s[i] = henc[(size_t)bt * INNERz + i];
    __syncthreads();

    // ---- build A tile in registers: ISA 16-bit A layout ----
    // lane (hi=0): elems 0..7 -> K=kb..kb+7, elems 8..15 -> K=kb+16..kb+23, kb=kc*32
    // lane (hi=1): same with kb=kc*32+8
    v16h areg[16];
    if (u < Uz + 1) {
        const float* hrow = hdec + ((size_t)(b * (Uz + 1) + u)) * INNERz;
        #pragma unroll
        for (int kc = 0; kc < 16; kc++) {
            int kb = kc * 32 + hi * 8;
            float xs[16];
            *(float4*)(xs + 0)  = *(const float4*)(hrow + kb);
            *(float4*)(xs + 4)  = *(const float4*)(hrow + kb + 4);
            *(float4*)(xs + 8)  = *(const float4*)(hrow + kb + 16);
            *(float4*)(xs + 12) = *(const float4*)(hrow + kb + 20);
            v16h a;
            #pragma unroll
            for (int j = 0; j < 8; j++) {
                a[j]     = (h16)FAST_TANH(henc_s[kb + j]      + xs[j]);
                a[8 + j] = (h16)FAST_TANH(henc_s[kb + 16 + j] + xs[8 + j]);
            }
            areg[kc] = a;
        }
    } else {
        #pragma unroll
        for (int kc = 0; kc < 16; kc++) {
            v16h a;
            #pragma unroll
            for (int j = 0; j < 16; j++) a[j] = (h16)0.0f;
            areg[kc] = a;
        }
    }

    // ---- online logsumexp over 63 v-tiles ----
    float mrun[8], srun[8];
    #pragma unroll
    for (int r = 0; r < 8; r++) { mrun[r] = -__builtin_inff(); srun[r] = 0.0f; }

    stage_tile_async(tid, W2T, 0, &Bst[0][0]);
    WAIT_ASYNC0();
    __syncthreads();

    for (int vt = 0; vt < 63; vt++) {
        if (vt + 1 < 63) stage_tile_async(tid, W2T, vt + 1, &Bst[(vt + 1) & 1][0]);

        // B layout: lane holds column N=m, K = kc*32 + hi*8 + 0..7, kc*32 + 16 + hi*8 + 0..7
        const h16* bp = &Bst[vt & 1][m * BSTR + hi * 8];
        float b2v = b2p[vt * 16 + m];

        v8f c = {};
        #pragma unroll
        for (int kc = 0; kc < 16; kc++) {
            v8h lo  = *(const v8h*)(bp + kc * 32);
            v8h hi8 = *(const v8h*)(bp + kc * 32 + 16);
            v16h bm = __builtin_shufflevector(lo, hi8, 0,1,2,3,4,5,6,7,8,9,10,11,12,13,14,15);
            c = __builtin_amdgcn_wmma_f32_16x16x32_f16(false, areg[kc], false, bm,
                                                       (short)0, c, false, false);
        }

        // per-row (C layout: VGPR r = row r + 8*hi, lanes-in-half = 16 columns)
        #pragma unroll
        for (int r = 0; r < 8; r++) {
            float x = c[r] + b2v;     // -inf on padded vocab columns
            float tm = x;
            tm = fmaxf(tm, __shfl_xor(tm, 1));
            tm = fmaxf(tm, __shfl_xor(tm, 2));
            tm = fmaxf(tm, __shfl_xor(tm, 4));
            tm = fmaxf(tm, __shfl_xor(tm, 8));
            float nm = fmaxf(mrun[r], tm);
            float e = __expf(x - nm);
            e += __shfl_xor(e, 1);
            e += __shfl_xor(e, 2);
            e += __shfl_xor(e, 4);
            e += __shfl_xor(e, 8);
            srun[r] = srun[r] * __expf(mrun[r] - nm) + e;
            mrun[r] = nm;
        }
        WAIT_ASYNC0();            // staged tile landed in LDS before publishing
        __syncthreads();
    }

    if (m == 0) {
        #pragma unroll
        for (int r = 0; r < 8; r++)
            lse_s[wave * 16 + hi * 8 + r] = mrun[r] + __logf(srun[r]);
    }
    __syncthreads();

    // ---- blank/label logits via f32 dot products from register-resident A ----
    int tok = (u < Uz) ? tokens[b * Uz + u] : 0;
    const h16* wb = W2T;                              // vocab row 0 (blank)
    const h16* wl = W2T + (size_t)tok * INNERz;
    float pb = 0.0f, pl = 0.0f;
    #pragma unroll
    for (int kc = 0; kc < 16; kc++) {
        int kb = kc * 32 + hi * 8;
        v16h a  = areg[kc];
        v8h b0 = *(const v8h*)(wb + kb);
        v8h b1v = *(const v8h*)(wb + kb + 16);
        v8h l0 = *(const v8h*)(wl + kb);
        v8h l1 = *(const v8h*)(wl + kb + 16);
        #pragma unroll
        for (int j = 0; j < 8; j++) {
            float a0 = (float)a[j], a1 = (float)a[8 + j];
            pb = fmaf(a0, (float)b0[j], fmaf(a1, (float)b1v[j], pb));
            pl = fmaf(a0, (float)l0[j], fmaf(a1, (float)l1[j], pl));
        }
    }
    pb += __shfl_xor(pb, 16);
    pl += __shfl_xor(pl, 16);
    if (u <= Uz) {
        float l = lse_s[u];
        size_t base = ((size_t)(b * Tz + t)) * LSTR;
        if (hi == 0)            blankO[base + u] = pb + b2p[0]   - l;
        if (hi == 1 && u < Uz)  labelO[base + u] = pl + b2p[tok] - l;
    }
}

// ---------------- kernel 3: anti-diagonal RNNT alpha DP + loss ----------------
__global__ __launch_bounds__(512)
void dp_kernel(const float* __restrict__ blankO, const float* __restrict__ labelO,
               const int* __restrict__ out_len, const int* __restrict__ tok_len,
               float* __restrict__ out)
{
    __shared__ float diag[2][Bz][UP];
    __shared__ float llv[Bz];
    const int tid = threadIdx.x;
    const int bb  = tid >> 7;       // 4 batches x 128 lanes
    const int u   = tid & 127;
    if (tid < Bz) llv[tid] = 0.0f;
    const int tI = out_len[bb] - 1;
    const int uI = tok_len[bb];
    const float* bl = blankO + (size_t)bb * Tz * LSTR;
    const float* la = labelO + (size_t)bb * Tz * LSTR;
    if (u == 0) {
        diag[0][bb][0] = 0.0f;                       // alpha(0,0)
        if (tI == 0 && uI == 0) llv[bb] = bl[0];
    }
    __syncthreads();
    for (int d = 1; d <= (Tz - 1) + Uz; d++) {
        const float* pv = diag[(d - 1) & 1][bb];
        float*       cv = diag[d & 1][bb];
        int t = d - u;
        bool valid = (u <= Uz) && (t >= 0) && (t <= Tz - 1);
        if (valid) {
            bool h1 = (t >= 1), h2 = (u >= 1);
            float p1 = h1 ? pv[u]     + bl[(size_t)(t - 1) * LSTR + u]       : 0.0f;
            float p2 = h2 ? pv[u - 1] + la[(size_t)t * LSTR + (u - 1)]      : 0.0f;
            float val;
            if (h1 && h2) {
                float mx = fmaxf(p1, p2);
                val = mx + __logf(__expf(p1 - mx) + __expf(p2 - mx));
            } else {
                val = h1 ? p1 : p2;
            }
            cv[u] = val;
            if (t == tI && u == uI) llv[bb] = val + bl[(size_t)t * LSTR + u];
        }
        __syncthreads();
    }
    if (tid == 0) out[0] = -0.25f * (llv[0] + llv[1] + llv[2] + llv[3]);
}

// ---------------- host launcher ----------------
extern "C" void kernel_launch(void* const* d_in, const int* in_sizes, int n_in,
                              void* d_out, int out_size, void* d_ws, size_t ws_size,
                              hipStream_t stream) {
    (void)in_sizes; (void)n_in; (void)out_size; (void)ws_size;
    const float* enc     = (const float*)d_in[0];
    const float* dec     = (const float*)d_in[1];
    const int*   tokens  = (const int*)d_in[2];
    const int*   outlen  = (const int*)d_in[3];
    const int*   toklen  = (const int*)d_in[4];
    const float* W1      = (const float*)d_in[5];
    const float* b1      = (const float*)d_in[6];
    const float* W2      = (const float*)d_in[7];
    const float* b2      = (const float*)d_in[8];
    float*       out     = (float*)d_out;

    size_t off = 0;
    char* base = (char*)d_ws;
    auto carve = [&](size_t bytes) {
        void* p = base + off;
        off += (bytes + 255) & ~(size_t)255;
        return p;
    };
    float* henc   = (float*)carve((size_t)Bz * Tz * INNERz * 4);        // 1.6 MB
    float* hdec   = (float*)carve((size_t)Bz * (Uz + 1) * INNERz * 4);  // 0.83 MB
    h16*   W2T    = (h16*)  carve((size_t)VP * INNERz * 2);             // 1.0 MB
    float* b2p    = (float*)carve((size_t)VP * 4);
    float* blankO = (float*)carve((size_t)Bz * Tz * LSTR * 4);          // 0.36 MB
    float* labelO = (float*)carve((size_t)Bz * Tz * LSTR * 4);          // 0.36 MB

    prep_kernel<<<(VP * INNERz + 255) / 256, 256, 0, stream>>>(W2, b2, W2T, b2p);
    row_gemm_kernel<<<Bz * Tz + Bz * (Uz + 1), 256, 0, stream>>>(enc, dec, W1, b1, henc, hdec);
    joint_kernel<<<Bz * Tz, 224, 0, stream>>>(henc, hdec, W2T, b2p, tokens, blankO, labelO);
    dp_kernel<<<1, 512, 0, stream>>>(blankO, labelO, outlen, toklen, out);
}
